// Model22_38388417691904
// MI455X (gfx1250) — matrix operand order repaired and checked by reading, not compiled, not run on hardware
//
#include <hip/hip_runtime.h>
#include <hip/hip_bf16.h>
#include <math.h>

#define HW   65536
#define IMG  256
#define NB   2
#define DIM  64
#define HID  170

typedef __attribute__((ext_vector_type(16))) _Float16 v16h;
typedef __attribute__((ext_vector_type(8)))  float    v8f;
typedef __attribute__((ext_vector_type(4)))  float    v4f;

// ---------------- weight staging: f32 [M,K] -> f16 [Mp,Kp] zero-padded ----------------
__global__ void k_stagew(const float* __restrict__ w, _Float16* __restrict__ o,
                         int M, int K, int Mp, int Kp) {
  int i = blockIdx.x * blockDim.x + threadIdx.x;
  if (i >= Mp * Kp) return;
  int r = i / Kp, c = i % Kp;
  float v = (r < M && c < K) ? w[r * K + c] : 0.f;
  o[i] = (_Float16)v;
}

// ---------------- channel LayerNorm: f32 NCHW -> f16 channel-last [pix][64] ----------------
__global__ void k_ln(const float* __restrict__ x, const float* __restrict__ g,
                     const float* __restrict__ b, _Float16* __restrict__ xn) {
  size_t gid = (size_t)blockIdx.x * blockDim.x + threadIdx.x;
  if (gid >= (size_t)NB * HW) return;
  int batch = (int)(gid >> 16);
  int p = (int)(gid & (HW - 1));
  const float* xb = x + (size_t)batch * DIM * HW + p;
  float v[DIM];
  float mu = 0.f;
  #pragma unroll
  for (int c = 0; c < DIM; ++c) { v[c] = xb[(size_t)c * HW]; mu += v[c]; }
  mu *= (1.f / DIM);
  float var = 0.f;
  #pragma unroll
  for (int c = 0; c < DIM; ++c) { float d = v[c] - mu; var += d * d; }
  var *= (1.f / DIM);
  float inv = rsqrtf(var + 1e-5f);
  _Float16* o = xn + gid * DIM;
  #pragma unroll
  for (int c = 0; c < DIM; ++c) o[c] = (_Float16)((v[c] - mu) * inv * g[c] + b[c]);
}

// ---------------- WMMA GEMM: Y[b][Mtot][HW] = W[Mtot,K] @ X[b][pix][K] (+res) ----------------
// A fragments (weights) preloaded to registers once per wave; loop over NT N-tiles with
// only 2x16B B-fragment loads per K-step. K compile-time -> constant instruction offsets.
template <int MTILES, int KTILES, bool ADD_RES>
__global__ void k_gemm(const _Float16* __restrict__ W,
                       const _Float16* __restrict__ X,
                       const float* __restrict__ res,
                       float* __restrict__ Y, int Mtot, int NT) {
  constexpr int K = KTILES * 32;
  int lane = threadIdx.x & 31;
  int wave = threadIdx.x >> 5;
  int nlo = lane & 15, hi = lane >> 4;
  int batch = blockIdx.z;
  int mOff = blockIdx.y * MTILES * 16;

  // Preload A fragments: lane = M row (mod 16); halfs [0..7] at +0, [8..15] at +16 halfs.
  v16h a[MTILES][KTILES];
  {
    const _Float16* wr0 = W + (size_t)(mOff + nlo) * K + hi * 8;
    #pragma unroll
    for (int mt = 0; mt < MTILES; ++mt)
      #pragma unroll
      for (int kt = 0; kt < KTILES; ++kt) {
        union { v16h h; v4f f[2]; } af;
        const _Float16* wr = wr0 + (size_t)mt * 16 * K + kt * 32;
        af.f[0] = *(const v4f*)(wr);
        af.f[1] = *(const v4f*)(wr + 16);
        a[mt][kt] = af.h;
      }
  }

  int ntile0 = (blockIdx.x * (blockDim.x >> 5) + wave) * NT;
  const _Float16* Xb = X + ((size_t)batch * HW + ntile0 * 16 + nlo) * K;
  for (int it = 0; it < NT; ++it) {
    int n0 = (ntile0 + it) * 16;
    // B fragments (K x 16): lane = N column; halfs [0..7] at K = kt*32 + hi*8, [8..15] at +16
    v16h b[KTILES];
    #pragma unroll
    for (int kt = 0; kt < KTILES; ++kt) {
      union { v16h h; v4f f[2]; } bf;
      const _Float16* bp = Xb + kt * 32 + hi * 8;
      bf.f[0] = *(const v4f*)(bp);
      bf.f[1] = *(const v4f*)(bp + 16);
      b[kt] = bf.h;
    }
    v8f acc[MTILES] = {};
    #pragma unroll
    for (int kt = 0; kt < KTILES; ++kt)
      #pragma unroll
      for (int mt = 0; mt < MTILES; ++mt)
        acc[mt] = __builtin_amdgcn_wmma_f32_16x16x32_f16(
            false, a[mt][kt], false, b[kt], (short)0, acc[mt], false, false);
    // D layout: VGPR r -> M = r + hi*8, N = nlo
    #pragma unroll
    for (int mt = 0; mt < MTILES; ++mt)
      #pragma unroll
      for (int r = 0; r < 8; ++r) {
        size_t idx = ((size_t)batch * Mtot + mOff + mt * 16 + hi * 8 + r) * (size_t)HW + n0 + nlo;
        float v = acc[mt][r];
        if (ADD_RES) v += res[idx];
        Y[idx] = v;
      }
    Xb += 16 * K;
  }
}

// ---------------- depthwise 3x3, fp32 NCHW, zero pad ----------------
__global__ void k_dwconv(const float* __restrict__ in, const float* __restrict__ wdw,
                         float* __restrict__ out, int C) {
  size_t gid = (size_t)blockIdx.x * blockDim.x + threadIdx.x;
  if (gid >= (size_t)NB * C * HW) return;
  int p = (int)(gid & (HW - 1));
  int rc = (int)(gid >> 16);       // batch*C + c
  int c = rc % C;
  int yy = p >> 8, xx = p & 255;
  const float* wp = wdw + c * 9;
  const float* ip = in + (gid - p); // channel-row base
  float acc = 0.f;
  #pragma unroll
  for (int dy = -1; dy <= 1; ++dy) {
    int y2 = yy + dy;
    if ((unsigned)y2 >= IMG) continue;
    #pragma unroll
    for (int dx = -1; dx <= 1; ++dx) {
      int x2 = xx + dx;
      if ((unsigned)x2 >= IMG) continue;
      acc += wp[(dy + 1) * 3 + dx + 1] * ip[y2 * IMG + x2];
    }
  }
  out[gid] = acc;
}

// ---------------- per-row L2 norms of q/k rows (rows 0..127 of qkv_dw per batch) ----------------
__global__ void k_norms(const float* __restrict__ qkv, float* __restrict__ norms) {
  int row = blockIdx.x;                  // 0..2*128-1
  int batch = row >> 7, r = row & 127;
  const float* p = qkv + ((size_t)batch * 192 + r) * HW;
  float s = 0.f;
  for (int i = threadIdx.x; i < HW; i += blockDim.x) { float v = p[i]; s += v * v; }
  __shared__ float red[256];
  red[threadIdx.x] = s; __syncthreads();
  for (int off = 128; off; off >>= 1) {
    if (threadIdx.x < off) red[threadIdx.x] += red[threadIdx.x + off];
    __syncthreads();
  }
  if (!threadIdx.x) norms[row] = fmaxf(sqrtf(red[0]), 1e-12f);
}

// ---------------- S[b,h,c,:] = sum_s q[c,s]*k[:,s]  (32 dots of length 65536 per block) ----------------
__global__ void k_qk(const float* __restrict__ qkv, float* __restrict__ S) {
  int rb = blockIdx.x;                   // 0..127 : batch | head | c
  int batch = rb >> 6, h = (rb >> 5) & 1, c = rb & 31;
  const float* q  = qkv + ((size_t)batch * 192 + h * 32 + c) * HW;
  const float* k0 = qkv + ((size_t)batch * 192 + 64 + h * 32) * HW;
  float acc[32] = {};
  for (int s = threadIdx.x; s < HW; s += blockDim.x) {
    float qv = q[s];
    #pragma unroll
    for (int d = 0; d < 32; ++d) acc[d] += qv * k0[(size_t)d * HW + s];
  }
  __shared__ float red[256];
  for (int d = 0; d < 32; ++d) {
    red[threadIdx.x] = acc[d]; __syncthreads();
    for (int off = 128; off; off >>= 1) {
      if (threadIdx.x < off) red[threadIdx.x] += red[threadIdx.x + off];
      __syncthreads();
    }
    if (!threadIdx.x) S[(size_t)rb * 32 + d] = red[0];
    __syncthreads();
  }
}

// ---------------- scale by 1/(nq*nk)*temp, row softmax (one wave32 per row) ----------------
__global__ void k_softmax(const float* __restrict__ S, const float* __restrict__ norms,
                          const float* __restrict__ temp, float* __restrict__ attn) {
  int row = blockIdx.x;                  // 128 rows
  int d = threadIdx.x;                   // 32 lanes
  int batch = row >> 6, h = (row >> 5) & 1, c = row & 31;
  float nq = norms[batch * 128 + h * 32 + c];
  float nk = norms[batch * 128 + 64 + h * 32 + d];
  float v = S[(size_t)row * 32 + d] / (nq * nk) * temp[h];
  float m = v;
  #pragma unroll
  for (int o = 16; o; o >>= 1) m = fmaxf(m, __shfl_xor(m, o, 32));
  float e = expf(v - m);
  float s = e;
  #pragma unroll
  for (int o = 16; o; o >>= 1) s += __shfl_xor(s, o, 32);
  attn[(size_t)row * 32 + d] = e / s;
}

// ---------------- out = attn @ v, written f16 channel-last [pix][64] ----------------
__global__ void k_av(const float* __restrict__ qkv, const float* __restrict__ attn,
                     _Float16* __restrict__ out) {
  size_t gid = (size_t)blockIdx.x * blockDim.x + threadIdx.x;
  if (gid >= (size_t)NB * HW) return;
  int batch = (int)(gid >> 16);
  int p = (int)(gid & (HW - 1));
  _Float16* op = out + gid * DIM;
  #pragma unroll
  for (int h = 0; h < 2; ++h) {
    const float* vb = qkv + ((size_t)batch * 192 + 128 + h * 32) * HW + p;
    float vv[32];
    #pragma unroll
    for (int d = 0; d < 32; ++d) vv[d] = vb[(size_t)d * HW];
    const float* at = attn + ((size_t)(batch * 2 + h)) * 32 * 32; // uniform -> scalar loads
    #pragma unroll 4
    for (int c = 0; c < 32; ++c) {
      float a = 0.f;
      #pragma unroll
      for (int d = 0; d < 32; ++d) a += at[c * 32 + d] * vv[d];
      op[h * 32 + c] = (_Float16)a;
    }
  }
}

// ---------------- fused GDFN dwconv3x3 + gelu(x1)*x2 -> f16 channel-last [pix][192] ----------------
__device__ __forceinline__ float geluf(float x) {
  return 0.5f * x * (1.f + erff(x * 0.70710678118654752f));
}

__global__ void k_gdfn(const float* __restrict__ pre, const float* __restrict__ wdw,
                       _Float16* __restrict__ gated) {
  size_t gid = (size_t)blockIdx.x * blockDim.x + threadIdx.x;
  if (gid >= (size_t)NB * HW) return;
  int batch = (int)(gid >> 16);
  int p = (int)(gid & (HW - 1));
  int yy = p >> 8, xx = p & 255;
  int offs[9]; float msk[9];
  {
    int t = 0;
    #pragma unroll
    for (int dy = -1; dy <= 1; ++dy)
      #pragma unroll
      for (int dx = -1; dx <= 1; ++dx) {
        int y2 = yy + dy, x2 = xx + dx;
        bool ok = ((unsigned)y2 < IMG) && ((unsigned)x2 < IMG);
        offs[t] = ok ? (y2 * IMG + x2) : 0;
        msk[t] = ok ? 1.f : 0.f;
        ++t;
      }
  }
  _Float16* op = gated + gid * 192;
  const float* base = pre + (size_t)batch * 384 * HW;   // w_in output padded to 384 rows
  #pragma unroll 2
  for (int c = 0; c < HID; ++c) {
    const float* w1 = wdw + c * 9;
    const float* w2 = wdw + (c + HID) * 9;
    const float* p1 = base + (size_t)c * HW;
    const float* p2 = base + (size_t)(c + HID) * HW;
    float a1 = 0.f, a2 = 0.f;
    #pragma unroll
    for (int t = 0; t < 9; ++t) {
      a1 += msk[t] * w1[t] * p1[offs[t]];
      a2 += msk[t] * w2[t] * p2[offs[t]];
    }
    op[c] = (_Float16)(geluf(a1) * a2);
  }
  #pragma unroll
  for (int c = HID; c < 192; ++c) op[c] = (_Float16)0.f;
}

// ============================== host side ==============================
extern "C" void kernel_launch(void* const* d_in, const int* in_sizes, int n_in,
                              void* d_out, int out_size, void* d_ws, size_t ws_size,
                              hipStream_t stream) {
  const float* x      = (const float*)d_in[0];
  const float* ln1_w  = (const float*)d_in[1];
  const float* ln1_b  = (const float*)d_in[2];
  const float* w_qkv  = (const float*)d_in[3];
  const float* dw_qkv = (const float*)d_in[4];
  const float* temp   = (const float*)d_in[5];
  const float* w_po   = (const float*)d_in[6];
  const float* ln2_w  = (const float*)d_in[7];
  const float* ln2_b  = (const float*)d_in[8];
  const float* w_in   = (const float*)d_in[9];
  const float* dw_ffn = (const float*)d_in[10];
  const float* w_out  = (const float*)d_in[11];
  float* out = (float*)d_out;

  // workspace layout (256B aligned); big buffers reused between attention and GDFN phases
  char* ws = (char*)d_ws;
  size_t off = 0;
  auto alloc = [&](size_t bytes) { size_t r = off; off += (bytes + 255) & ~(size_t)255; return r; };
  const size_t o_xn   = alloc((size_t)NB * HW * DIM * 2);     // f16 LN output, channel-last
  const size_t o_big1 = alloc((size_t)NB * 384 * HW * 4);     // qkv f32 / ffn-pre f32 (384 rows)
  const size_t o_big2 = alloc((size_t)NB * 192 * HW * 4);     // qkv_dw f32 / gated f16
  const size_t o_aout = alloc((size_t)NB * HW * DIM * 2);     // attn output f16, channel-last
  const size_t o_y1   = alloc((size_t)NB * DIM * HW * 4);     // x + attention (residual)
  const size_t o_S    = alloc(4096 * 4);
  const size_t o_attn = alloc(4096 * 4);
  const size_t o_nrm  = alloc(256 * 4);
  const size_t o_wq   = alloc(192 * 64 * 2);
  const size_t o_wpo  = alloc(64 * 64 * 2);
  const size_t o_win  = alloc(384 * 64 * 2);
  const size_t o_wout = alloc(64 * 192 * 2);

  _Float16* xn    = (_Float16*)(ws + o_xn);
  float*    big1  = (float*)(ws + o_big1);
  float*    big2  = (float*)(ws + o_big2);
  _Float16* aout  = (_Float16*)(ws + o_aout);
  float*    y1    = (float*)(ws + o_y1);
  float*    Sbuf  = (float*)(ws + o_S);
  float*    attn  = (float*)(ws + o_attn);
  float*    nrm   = (float*)(ws + o_nrm);
  _Float16* wqf   = (_Float16*)(ws + o_wq);
  _Float16* wpof  = (_Float16*)(ws + o_wpo);
  _Float16* winf  = (_Float16*)(ws + o_win);
  _Float16* woutf = (_Float16*)(ws + o_wout);

  // stage weights to f16 (padded)
  k_stagew<<<(192 * 64 + 255) / 256, 256, 0, stream>>>(w_qkv, wqf, 192, 64, 192, 64);
  k_stagew<<<(64 * 64 + 255) / 256, 256, 0, stream>>>(w_po, wpof, 64, 64, 64, 64);
  k_stagew<<<(384 * 64 + 255) / 256, 256, 0, stream>>>(w_in, winf, 340, 64, 384, 64);
  k_stagew<<<(64 * 192 + 255) / 256, 256, 0, stream>>>(w_out, woutf, 64, 170, 64, 192);

  const int LN_BLOCKS = (NB * HW) / 256;                 // 512
  const int NT = 4;                                      // N tiles per wave
  const int GX = HW / (16 * 8 * NT);                     // 128 blocks (8 waves/block)
  // ---- attention branch ----
  k_ln<<<LN_BLOCKS, 256, 0, stream>>>(x, ln1_w, ln1_b, xn);
  k_gemm<4, 2, false><<<dim3(GX, 3, NB), 256, 0, stream>>>(wqf, xn, nullptr, big1, 192, NT);
  k_dwconv<<<(NB * 192 * HW) / 256, 256, 0, stream>>>(big1, dw_qkv, big2, 192);
  k_norms<<<NB * 128, 256, 0, stream>>>(big2, nrm);
  k_qk<<<NB * 64, 256, 0, stream>>>(big2, Sbuf);
  k_softmax<<<NB * 64, 32, 0, stream>>>(Sbuf, nrm, temp, attn);
  k_av<<<LN_BLOCKS, 256, 0, stream>>>(big2, attn, aout);
  k_gemm<4, 2, true><<<dim3(GX, 1, NB), 256, 0, stream>>>(wpof, aout, x, y1, 64, NT);
  // ---- GDFN branch ----
  k_ln<<<LN_BLOCKS, 256, 0, stream>>>(y1, ln2_w, ln2_b, xn);
  k_gemm<4, 2, false><<<dim3(GX, 6, NB), 256, 0, stream>>>(winf, xn, nullptr, big1, 384, NT);
  k_gdfn<<<LN_BLOCKS, 256, 0, stream>>>(big1, dw_ffn, (_Float16*)big2);
  k_gemm<2, 6, true><<<dim3(GX, 2, NB), 256, 0, stream>>>(woutf, (_Float16*)big2, y1, out, 64, NT);

  (void)in_sizes; (void)n_in; (void)out_size; (void)ws_size;
}
